// BoundaryPredictor2_70068096467864
// MI455X (gfx1250) — compile-verified, matrix operand order checked by reading, not run on hardware
//
#include <hip/hip_runtime.h>
#include <math.h>

// ---------------- types for WMMA fragments ----------------
typedef __bf16  v16bf __attribute__((ext_vector_type(16)));
typedef float   v8f   __attribute__((ext_vector_type(8)));
typedef unsigned int u32x4 __attribute__((ext_vector_type(4)));

union Frag16 { v16bf v; u32x4 q[2]; };

#define WMMA_BF16(a, b, c) \
  __builtin_amdgcn_wmma_f32_16x16x32_bf16(false, (a), false, (b), (short)0, (c), false, false)

// gfx1250 async memory->LDS DMA (ASYNCcnt-tracked; no VGPR round trip).
// LDS byte offset == low 32 bits of the generic pointer (flat->LDS aperture map).
__device__ __forceinline__ void async_copy_b128(void* lds_dst, const void* gsrc) {
  unsigned ldsa = (unsigned)(unsigned long long)lds_dst;
  unsigned long long ga = (unsigned long long)gsrc;
  asm volatile("global_load_async_to_lds_b128 %0, %1, off"
               :: "v"(ldsa), "v"(ga) : "memory");
}
#define WAIT_ASYNC_ALL() asm volatile("s_wait_asynccnt 0" ::: "memory")
#define WAIT_ASYNC_3()   asm volatile("s_wait_asynccnt 3" ::: "memory")
#define WAIT_ASYNC_4()   asm volatile("s_wait_asynccnt 4" ::: "memory")

// f32 -> bf16 round-to-nearest-even
__device__ __forceinline__ unsigned short f2bf(float f) {
  unsigned int u = __float_as_uint(f);
  u = u + 0x7FFFu + ((u >> 16) & 1u);
  return (unsigned short)(u >> 16);
}

// Load one 16x32 (MxK) bf16 A/B fragment row per the CDNA5 ISA layout:
// lane 0-15 : row = lane,    K = {hf*8..hf*8+7} U {16+hf*8..16+hf*8+7}, hf=0
// lane 16-31: row = lane-16, same with hf=1
__device__ __forceinline__ v16bf load_frag(const unsigned short* rowbase, int hf) {
  Frag16 f;
  f.q[0] = *(const u32x4*)(rowbase + hf * 8);
  f.q[1] = *(const u32x4*)(rowbase + 16 + hf * 8);
  return f.v;
}

// ---------------- problem constants ----------------
#define BB 4
#define LL 2048
#define DD 512
#define HH 8
#define HD 64
#define SCALE 0.125f   // 64^-0.5
#define PRIOR 0.2f

// ======================================================================
// Kernel 0a: f32 -> bf16 convert (grid-stride)
// ======================================================================
__global__ void cvt_bf16_kernel(const float* __restrict__ s,
                                unsigned short* __restrict__ d, int n) {
  int i = blockIdx.x * blockDim.x + threadIdx.x;
  int stride = gridDim.x * blockDim.x;
  for (; i < n; i += stride) d[i] = f2bf(s[i]);
}

// Kernel 0b: zero a float buffer
__global__ void zero_kernel(float* __restrict__ p, int n) {
  int i = blockIdx.x * blockDim.x + threadIdx.x;
  if (i < n) p[i] = 0.0f;
}

// ======================================================================
// Kernel 1: fused QKV projection  C = X * W^T   (M=8192, N=512, K=512)
// blockIdx.z: 0->Q, 1->K (layout [B,H,L,HD]), 2->V stored transposed [B,H,HD,L]
// Block: 256 threads (8 waves). Tile: 128(M) x 64(N). Each wave: 16x64 strip.
// Double-buffered async-to-LDS staging; K-loop unrolled x2 (static buf index);
// all fragments loaded before the 4 back-to-back WMMAs.
// ======================================================================
__global__ __launch_bounds__(256) void qkv_gemm(
    const unsigned short* __restrict__ X,
    const unsigned short* __restrict__ Wq,
    const unsigned short* __restrict__ Wk,
    const unsigned short* __restrict__ Wv,
    unsigned short* __restrict__ Qb,
    unsigned short* __restrict__ Kb,
    unsigned short* __restrict__ Vtb) {
  __shared__ unsigned short At[2][128][32];
  __shared__ unsigned short Bt[2][64][32];

  const int t    = threadIdx.x;
  const int wv   = t >> 5;
  const int lane = t & 31;
  const int hf   = lane >> 4;
  const int l15  = lane & 15;
  const int which = blockIdx.z;
  const unsigned short* W = (which == 0) ? Wq : (which == 1) ? Wk : Wv;
  const int m0 = blockIdx.x * 128;
  const int n0 = blockIdx.y * 64;

  // per-thread static chunk assignments (uniform -> straight line)
  const int ra0 = t >> 2,          sa0 = t & 3;            // A chunk 0
  const int ra1 = (t + 256) >> 2,  sa1 = (t + 256) & 3;    // A chunk 1
  const int rb  = t >> 2,          sb  = t & 3;            // B chunk

  auto stage = [&](int buf, int kk) {
    async_copy_b128(&At[buf][ra0][sa0 * 8],
                    X + (size_t)(m0 + ra0) * DD + kk + sa0 * 8);
    async_copy_b128(&At[buf][ra1][sa1 * 8],
                    X + (size_t)(m0 + ra1) * DD + kk + sa1 * 8);
    async_copy_b128(&Bt[buf][rb][sb * 8],
                    W + (size_t)(n0 + rb) * DD + kk + sb * 8);
  };

  v8f acc[4] = {};

  auto step = [&](int buf, int kk) {   // buf is a literal at each call site
    const bool more = (kk + 32) < DD;
    if (more) stage(buf ^ 1, kk + 32);          // prefetch next tile (async)
    if (more) WAIT_ASYNC_3(); else WAIT_ASYNC_ALL();
    __syncthreads();

    // load ALL fragments first -> single dscnt wait, back-to-back WMMAs
    v16bf a  = load_frag(&At[buf][wv * 16 + l15][0], hf);
    v16bf b0 = load_frag(&Bt[buf][0  + l15][0], hf);
    v16bf b1 = load_frag(&Bt[buf][16 + l15][0], hf);
    v16bf b2 = load_frag(&Bt[buf][32 + l15][0], hf);
    v16bf b3 = load_frag(&Bt[buf][48 + l15][0], hf);
    acc[0] = WMMA_BF16(a, b0, acc[0]);
    acc[1] = WMMA_BF16(a, b1, acc[1]);
    acc[2] = WMMA_BF16(a, b2, acc[2]);
    acc[3] = WMMA_BF16(a, b3, acc[3]);
    __syncthreads();   // all reads done before this buffer is re-filled
  };

  stage(0, 0);
  for (int kk = 0; kk < DD; kk += 64) {   // 8 double-steps, static buf ids
    step(0, kk);
    step(1, kk + 32);
  }

  // store: C element (row = j + 8*hf, col = l15) in VGPR j
#pragma unroll
  for (int n = 0; n < 4; n++) {
    int gn = n0 + n * 16 + l15;
    int h = gn >> 6, hd = gn & 63;
#pragma unroll
    for (int j = 0; j < 8; j++) {
      int gm = m0 + wv * 16 + j + hf * 8;
      int b = gm >> 11, l = gm & 2047;
      unsigned short val = f2bf(acc[n][j]);
      if (which == 2) {
        Vtb[(((size_t)(b * HH + h) * HD + hd) << 11) + l] = val;
      } else {
        size_t idx = ((((size_t)(b * HH + h)) << 11) + l) * HD + hd;
        if (which == 0) Qb[idx] = val; else Kb[idx] = val;
      }
    }
  }
}

// ======================================================================
// Kernel 2: flash attention + fused (att . Wo) -> atomicAdd into logits.
// Grid: (L/64, B*H). Block: 128 threads (4 waves, 16 queries each).
// Never materializes att (only boundary_logits needs it).
// Double-buffered async-to-LDS staging; key loop unrolled x2.
// ======================================================================
__global__ __launch_bounds__(128) void flash_attn(
    const unsigned short* __restrict__ Qb,
    const unsigned short* __restrict__ Kb,
    const unsigned short* __restrict__ Vtb,
    const float* __restrict__ Wo,
    float* __restrict__ logits) {
  __shared__ unsigned short Kt[2][32][64];    // [key][hd]
  __shared__ unsigned short Vt[2][64][32];    // [hd][key] (V pre-transposed)
  __shared__ unsigned short Pt[4][16][32];    // per-wave P tile [q][key]

  const int t    = threadIdx.x;
  const int wv   = t >> 5;
  const int lane = t & 31;
  const int hf   = lane >> 4;
  const int l15  = lane & 15;
  const int bh   = blockIdx.y;                // b*H + h
  const int h    = bh & (HH - 1);
  const int q0   = blockIdx.x * 64;
  const size_t base = (size_t)bh * (LL * HD);

  // static chunk assignments for staging
  const int rk0 = t >> 3,          sk0 = t & 7;            // Kt chunk 0
  const int rk1 = (t + 128) >> 3,  sk1 = (t + 128) & 7;    // Kt chunk 1
  const int rv0 = t >> 2,          sv0 = t & 3;            // Vt chunk 0
  const int rv1 = (t + 128) >> 2,  sv1 = (t + 128) & 3;    // Vt chunk 1

  auto stage = [&](int buf, int kb) {
    async_copy_b128(&Kt[buf][rk0][sk0 * 8],
                    Kb + base + (size_t)(kb + rk0) * HD + sk0 * 8);
    async_copy_b128(&Kt[buf][rk1][sk1 * 8],
                    Kb + base + (size_t)(kb + rk1) * HD + sk1 * 8);
    async_copy_b128(&Vt[buf][rv0][sv0 * 8],
                    Vtb + base + ((size_t)rv0 << 11) + kb + sv0 * 8);
    async_copy_b128(&Vt[buf][rv1][sv1 * 8],
                    Vtb + base + ((size_t)rv1 << 11) + kb + sv1 * 8);
  };

  // Q fragments for this wave's 16 query rows (K-dim 64 -> 2 fragments)
  const unsigned short* qptr = Qb + base + (size_t)(q0 + wv * 16 + l15) * HD;
  v16bf qa0 = load_frag(qptr, hf);
  v16bf qa1 = load_frag(qptr + 32, hf);

  float mrow[8], lrow[8];
  v8f O[4] = {};
#pragma unroll
  for (int j = 0; j < 8; j++) { mrow[j] = -3.0e38f; lrow[j] = 0.0f; }

  auto step = [&](int buf, int kb) {   // buf is a literal at each call site
    const bool more = (kb + 32) < LL;
    if (more) stage(buf ^ 1, kb + 32);          // prefetch next K/V^T block
    if (more) WAIT_ASYNC_4(); else WAIT_ASYNC_ALL();
    __syncthreads();

    // S = Q K^T : fragments first, then 4 back-to-back WMMAs
    v16bf b0 = load_frag(&Kt[buf][l15][0],       hf);
    v16bf b1 = load_frag(&Kt[buf][l15][32],      hf);
    v16bf c0 = load_frag(&Kt[buf][16 + l15][0],  hf);
    v16bf c1 = load_frag(&Kt[buf][16 + l15][32], hf);
    v8f s0 = {}, s1 = {};
    s0 = WMMA_BF16(qa0, b0, s0);
    s1 = WMMA_BF16(qa0, c0, s1);
    s0 = WMMA_BF16(qa1, b1, s0);
    s1 = WMMA_BF16(qa1, c1, s1);

    // online softmax; row stats via 16-lane shfl_xor reductions (stay in half)
    float p0r[8], p1r[8];
#pragma unroll
    for (int j = 0; j < 8; j++) {
      float a0 = s0[j] * SCALE, a1 = s1[j] * SCALE;
      float mx = fmaxf(a0, a1);
#pragma unroll
      for (int d = 1; d < 16; d <<= 1) mx = fmaxf(mx, __shfl_xor(mx, d, 32));
      float mn = fmaxf(mrow[j], mx);
      float rsc = __expf(mrow[j] - mn);
      float p0 = __expf(a0 - mn), p1 = __expf(a1 - mn);
      float rs = p0 + p1;
#pragma unroll
      for (int d = 1; d < 16; d <<= 1) rs += __shfl_xor(rs, d, 32);
      lrow[j] = lrow[j] * rsc + rs;
      mrow[j] = mn;
      p0r[j] = p0; p1r[j] = p1;
#pragma unroll
      for (int n = 0; n < 4; n++) O[n][j] *= rsc;
    }

    // C-layout -> A-layout conversion through LDS
#pragma unroll
    for (int j = 0; j < 8; j++) {
      Pt[wv][j + hf * 8][l15]      = f2bf(p0r[j]);
      Pt[wv][j + hf * 8][16 + l15] = f2bf(p1r[j]);
    }
    __syncthreads();

    // P x V : fragments first, then 4 back-to-back WMMAs
    v16bf pa  = load_frag(&Pt[wv][l15][0], hf);
    v16bf vb0 = load_frag(&Vt[buf][0  + l15][0], hf);
    v16bf vb1 = load_frag(&Vt[buf][16 + l15][0], hf);
    v16bf vb2 = load_frag(&Vt[buf][32 + l15][0], hf);
    v16bf vb3 = load_frag(&Vt[buf][48 + l15][0], hf);
    O[0] = WMMA_BF16(pa, vb0, O[0]);
    O[1] = WMMA_BF16(pa, vb1, O[1]);
    O[2] = WMMA_BF16(pa, vb2, O[2]);
    O[3] = WMMA_BF16(pa, vb3, O[3]);
    __syncthreads();   // all reads done before buffers are re-filled
  };

  stage(0, 0);
  for (int kb = 0; kb < LL; kb += 64) {   // 32 double-steps, static buf ids
    step(0, kb);
    step(1, kb + 32);
  }

  // epilogue: boundary_logits contribution = sum_hd (O[q][hd]/l[q]) * Wo[h*64+hd]
  float wov[4];
#pragma unroll
  for (int n = 0; n < 4; n++) wov[n] = Wo[h * HD + n * 16 + l15];

  float accj[8];
#pragma unroll
  for (int j = 0; j < 8; j++) {
    float inv = 1.0f / lrow[j];
    float s = 0.0f;
#pragma unroll
    for (int n = 0; n < 4; n++) s += O[n][j] * inv * wov[n];
#pragma unroll
    for (int d = 1; d < 16; d <<= 1) s += __shfl_xor(s, d, 32);
    accj[j] = s;
  }
  if (l15 == 0) {
    int b = bh >> 3;
#pragma unroll
    for (int j = 0; j < 8; j++)
      atomicAdd(&logits[b * LL + q0 + wv * 16 + j + hf * 8], accj[j]);
  }
}

// ======================================================================
// Kernel 3: boundary decision + per-batch segment-id scan.
// boundary_scores == 1 exactly (softmax rows sum to 1); logit(sigmoid(x)) == x.
// One block (256 threads) per batch; each thread owns 8 tokens.
// ======================================================================
__global__ __launch_bounds__(256) void boundary_scan(
    const float* __restrict__ logits, const float* __restrict__ u,
    int* __restrict__ ids, int* __restrict__ segs,
    float* __restrict__ kcnt, int* __restrict__ nseg) {
  const int b = blockIdx.x, t = threadIdx.x;
  __shared__ float part[256];
  __shared__ unsigned char hs[LL];

  float hv[8];
  float sum = 0.0f;
#pragma unroll
  for (int i = 0; i < 8; i++) {
    int l = t * 8 + i;
    float x  = logits[b * LL + l] + 1.0f;           // + boundary_scores (==1)
    float uu = u[b * LL + l];
    float noise = logf(uu) - log1pf(-uu);
    float soft  = 1.0f / (1.0f + __expf(-(x + noise)));
    float hd = (soft > 0.5f) ? 1.0f : 0.0f;
    hv[i] = hd; sum += hd;
    hs[l] = (unsigned char)hd;
  }
  part[t] = sum;
  __syncthreads();
  for (int off = 1; off < 256; off <<= 1) {
    float v = (t >= off) ? part[t - off] : 0.0f;
    __syncthreads();
    part[t] += v;
    __syncthreads();
  }
  float run = part[t] - sum;   // exclusive prefix
#pragma unroll
  for (int i = 0; i < 8; i++) {
    int l = t * 8 + i;
    int id = (int)run;
    ids[b * LL + l] = id;
    // segment start iff l==0 or hard[l-1]==1 (ids are non-decreasing)
    if (l == 0 || hs[l - 1]) segs[b * (LL + 1) + id] = l;
    run += hv[i];
  }
  if (t == 255) {
    float total = part[255];
    int ns = (int)(total - hv[7]) + 1;   // ids[L-1] + 1
    segs[b * (LL + 1) + ns] = LL;        // sentinel
    kcnt[b] = total;
    nseg[b] = ns;
  }
}

// ======================================================================
// Kernel 4: segment-mean pooling. One block per output row (b,j).
// Rows j >= num_segments are zero (matches 0/max(c,1) in reference).
// ======================================================================
__global__ __launch_bounds__(128) void pool_kernel(
    const float* __restrict__ hidden, const int* __restrict__ segs,
    const int* __restrict__ nseg, float* __restrict__ out) {
  int bid = blockIdx.x;
  int b = bid >> 11, j = bid & 2047;
  int t = threadIdx.x;
  float a0 = 0, a1 = 0, a2 = 0, a3 = 0;
  int ns = nseg[b];
  if (j < ns) {
    int s0 = segs[b * (LL + 1) + j], s1 = segs[b * (LL + 1) + j + 1];
    const float* hp = hidden + ((size_t)b << 11) * DD + t * 4;
    for (int l = s0; l < s1; l++) {
      const float4 v = *(const float4*)(hp + (size_t)l * DD);
      a0 += v.x; a1 += v.y; a2 += v.z; a3 += v.w;
    }
    int c = s1 - s0; if (c < 1) c = 1;
    float inv = 1.0f / (float)c;
    a0 *= inv; a1 *= inv; a2 *= inv; a3 *= inv;
  }
  float4 r; r.x = a0; r.y = a1; r.z = a2; r.w = a3;
  *(float4*)(out + (((size_t)b << 11) + j) * DD + t * 4) = r;
}

// ======================================================================
// Kernel 5: binomial log-prob loss
// ======================================================================
__global__ void loss_kernel(const float* __restrict__ kcnt,
                            float* __restrict__ out) {
  int t = threadIdx.x;
  float term = 0.0f;
  if (t < BB) {
    float k = kcnt[t];
    float n = (float)LL;
    term = lgammaf(n + 1.0f) - lgammaf(k + 1.0f) - lgammaf(n - k + 1.0f)
         + k * logf(PRIOR) + (n - k) * log1pf(-PRIOR);
  }
#pragma unroll
  for (int d = 1; d < 32; d <<= 1) term += __shfl_xor(term, d, 32);
  if (t == 0) out[(size_t)BB * LL * DD] = -(term / (float)BB) / (float)LL;
}

// ======================================================================
extern "C" void kernel_launch(void* const* d_in, const int* in_sizes, int n_in,
                              void* d_out, int out_size, void* d_ws, size_t ws_size,
                              hipStream_t stream) {
  const float* hidden = (const float*)d_in[0];
  const float* Wq = (const float*)d_in[1];
  const float* Wk = (const float*)d_in[2];
  const float* Wv = (const float*)d_in[3];
  const float* Wo = (const float*)d_in[4];
  const float* u  = (const float*)d_in[5];
  float* out = (float*)d_out;

  char* w = (char*)d_ws;
  size_t off = 0;
  auto walloc = [&](size_t bytes) -> void* {
    void* p = w + off;
    off += (bytes + 255) & ~(size_t)255;
    return p;
  };
  const size_t ML = (size_t)BB * LL;           // 8192
  unsigned short* Xb  = (unsigned short*)walloc(ML * DD * 2);   // 8 MB
  unsigned short* Wqb = (unsigned short*)walloc((size_t)DD * DD * 2);
  unsigned short* Wkb = (unsigned short*)walloc((size_t)DD * DD * 2);
  unsigned short* Wvb = (unsigned short*)walloc((size_t)DD * DD * 2);
  unsigned short* Qb  = (unsigned short*)walloc(ML * DD * 2);   // [B,H,L,HD]
  unsigned short* Kb  = (unsigned short*)walloc(ML * DD * 2);   // [B,H,L,HD]
  unsigned short* Vtb = (unsigned short*)walloc(ML * DD * 2);   // [B,H,HD,L]
  float* logits = (float*)walloc(ML * 4);
  int*   ids    = (int*)walloc(ML * 4);
  int*   segs   = (int*)walloc((size_t)BB * (LL + 1) * 4);
  float* kcnt   = (float*)walloc(BB * 4);
  int*   nseg   = (int*)walloc(BB * 4);
  (void)ws_size; (void)in_sizes; (void)n_in; (void)out_size;

  // 0) converts + zero
  cvt_bf16_kernel<<<2048, 256, 0, stream>>>(hidden, Xb, (int)(ML * DD));
  cvt_bf16_kernel<<<256, 256, 0, stream>>>(Wq, Wqb, DD * DD);
  cvt_bf16_kernel<<<256, 256, 0, stream>>>(Wk, Wkb, DD * DD);
  cvt_bf16_kernel<<<256, 256, 0, stream>>>(Wv, Wvb, DD * DD);
  zero_kernel<<<32, 256, 0, stream>>>(logits, (int)ML);

  // 1) QKV projections (z = 0:Q, 1:K, 2:V-transposed)
  qkv_gemm<<<dim3(64, 8, 3), 256, 0, stream>>>(Xb, Wqb, Wkb, Wvb, Qb, Kb, Vtb);

  // 2) flash attention + fused att.Wo -> logits
  flash_attn<<<dim3(LL / 64, BB * HH), 128, 0, stream>>>(Qb, Kb, Vtb, Wo, logits);

  // 3) boundary decision + segment scan
  boundary_scan<<<BB, 256, 0, stream>>>(logits, u, ids, segs, kcnt, nseg);

  // 4) segment-mean pooling -> out[0 .. B*L*D)
  pool_kernel<<<BB * LL, 128, 0, stream>>>(hidden, segs, nseg, out);

  // 5) loss -> out[B*L*D]
  loss_kernel<<<1, 32, 0, stream>>>(kcnt, out);
}
